// CorrelationLayer_62208306315343
// MI455X (gfx1250) — matrix operand order repaired and checked by reading, not compiled, not run on hardware
//
#include <hip/hip_runtime.h>
#include <stdint.h>

// Problem constants (match reference setup_inputs)
#define Bn   16
#define Cn   256
#define Hn   128
#define Wn   128
#define NSH  81           // (2*4+1)^2 displacements

// Padded f2 layout: rows -4..131 -> +4, cols -4..139 -> +4
#define PHn  (Hn + 8)     // 136
#define PWn  (Wn + 16)    // 144

typedef __attribute__((ext_vector_type(16))) __bf16 v16bf;
typedef __attribute__((ext_vector_type(8)))  __bf16 v8bf;
typedef __attribute__((ext_vector_type(8)))  float  v8f;

// ---------------------------------------------------------------------------
// Zero-fill the padded g2 array (borders must read as 0.0bf16).
// ---------------------------------------------------------------------------
__global__ __launch_bounds__(256)
void zero_ws_kernel(uint4* __restrict__ p, size_t n16) {
  const size_t i = (size_t)blockIdx.x * 256 + threadIdx.x;
  if (i < n16) {
    uint4 z; z.x = 0u; z.y = 0u; z.z = 0u; z.w = 0u;
    p[i] = z;
  }
}

// ---------------------------------------------------------------------------
// Pass 1: L2-normalize over C, write bf16 pixel-major.
//   feat1 -> g1 [B][H][W][C]           (unpadded)
//   feat2 -> g2 [B][PH][PW][C]         (interior of zeroed padded array)
// One block = one 32-pixel strip of one row. Coalesced f32 reads (w fastest),
// LDS transpose, coalesced bf16 writes (c fastest).
// ---------------------------------------------------------------------------
__global__ __launch_bounds__(256)
void norm_bf16_kernel(const float* __restrict__ f1,
                      const float* __restrict__ f2,
                      uint16_t* __restrict__ g1,
                      uint16_t* __restrict__ g2) {
  __shared__ float tile[32][Cn + 1];   // +1 pad: conflict-free transpose
  __shared__ float part[8][32];
  __shared__ float scaleS[32];

  const int tx = threadIdx.x & 31;     // pixel within strip
  const int ty = threadIdx.x >> 5;     // channel phase (0..7)
  const int w0 = blockIdx.x * 32;
  const int h  = blockIdx.y;
  const int z  = blockIdx.z;           // 0..15 -> feat1 b, 16..31 -> feat2 b
  const int b  = z & 15;

  const float* src = (z < Bn) ? f1 : f2;
  const float* p = src + (((size_t)b * Cn) * Hn + h) * Wn + (w0 + tx);
  const size_t chStride = (size_t)Hn * Wn;

  float acc = 0.0f;
#pragma unroll 4
  for (int j = 0; j < 32; ++j) {
    const int c = ty + 8 * j;
    const float v = p[(size_t)c * chStride];
    tile[tx][c] = v;
    acc += v * v;
  }
  part[ty][tx] = acc;
  __syncthreads();

  if (ty == 0) {
    float s = 0.0f;
#pragma unroll
    for (int r = 0; r < 8; ++r) s += part[r][tx];
    const float nrm = sqrtf(s);
    scaleS[tx] = 1.0f / fmaxf(nrm, 1e-12f);
  }
  __syncthreads();

  uint16_t* q;
  if (z < Bn) {
    q = g1 + ((((size_t)b * Hn) + h) * Wn + w0) * Cn;
  } else {
    q = g2 + ((((size_t)b * PHn) + (h + 4)) * PWn + (w0 + 4)) * Cn;
  }

#pragma unroll 4
  for (int it = 0; it < 32; ++it) {
    const int pix = it;
    const int c   = threadIdx.x;               // lanes -> consecutive c
    const float v = tile[pix][c] * scaleS[pix];
    // round-to-nearest-even f32 -> bf16 (integer path)
    const uint32_t u = __float_as_uint(v);
    const uint32_t r = (u + 0x7FFFu + ((u >> 16) & 1u)) >> 16;
    q[(size_t)pix * Cn + c] = (uint16_t)r;
  }
}

// ---------------------------------------------------------------------------
// Pass 2: banded correlation via v_wmma_f32_16x16x32_bf16.
// Block = 9 waves (288 threads); wave wv handles dy = wv-4 for a 16-pixel
// strip at (b, h, w0). Two 16x16 D tiles per wave, K=256 in 8 WMMA steps.
// All B addresses are in-bounds thanks to the zero-padded g2 -> the inner
// loop is pure loads + WMMA (no EXEC masking, no zero fills).
// ---------------------------------------------------------------------------
__global__ __launch_bounds__(288)
void corr_wmma_kernel(const uint16_t* __restrict__ g1,
                      const uint16_t* __restrict__ g2,
                      float* __restrict__ out) {
  const int lane = threadIdx.x & 31;
  const int wv   = threadIdx.x >> 5;       // 0..8 -> dy = wv-4
  const int w0   = blockIdx.x * 16;
  const int h    = blockIdx.y;
  const int b    = blockIdx.z;

  const int m    = lane & 15;
  const int half = lane >> 4;

  // padded row index: ry + 4 = h - (wv-4) + 4
  const int pry = h - wv + 8;              // in [0, 135]

  v8f c0 = {};
  v8f c1 = {};

  // A: f1n pixel (w0+m); per-lane 16 contiguous K starting at half*16
  // (same K order as B).
  const uint16_t* pA = g1 + ((((size_t)b * Hn) + h) * Wn + (w0 + m)) * Cn + half * 16;

  // B: padded f2n row pry; tile0 col = w0+m (padded), tile1 col = +16;
  // per-lane 16 contiguous K starting at half*16 (same K order as A).
  const uint16_t* pB0 = g2 + ((((size_t)b * PHn) + pry) * PWn + (w0 + m)) * Cn + half * 16;
  const uint16_t* pB1 = pB0 + (size_t)16 * Cn;

#pragma unroll
  for (int kk = 0; kk < 8; ++kk) {
    const int k0 = kk * 32;

    const v8bf alo = *(const v8bf*)(pA + k0);
    const v8bf ahi = *(const v8bf*)(pA + k0 + 8);
    const v16bf a = __builtin_shufflevector(alo, ahi,
        0, 1, 2, 3, 4, 5, 6, 7, 8, 9, 10, 11, 12, 13, 14, 15);

    const v16bf b0 = *(const v16bf*)(pB0 + k0);
    const v16bf b1 = *(const v16bf*)(pB1 + k0);

    c0 = __builtin_amdgcn_wmma_f32_16x16x32_bf16(
        false, a, false, b0, (short)0, c0, false, false);
    c1 = __builtin_amdgcn_wmma_f32_16x16x32_bf16(
        false, a, false, b1, (short)0, c1, false, false);
  }

  // Scatter the valid band of both tiles.
  // D layout: lane L, VGPR r -> D[mm = r + (L>=16)*8, n = L&15].
#pragma unroll
  for (int r = 0; r < 8; ++r) {
    const int mm = r + half * 8;
    const int n  = m;
    const size_t obase = (((size_t)b * NSH) * Hn + h) * Wn + (w0 + mm);

    const int dx0 = mm + 4 - n;           // tile0 column pixel = w0-4+n
    if (dx0 >= -4 && dx0 <= 4) {
      const int s = wv * 9 + (dx0 + 4);
      out[obase + (size_t)s * Hn * Wn] = c0[r];
    }
    const int dx1 = mm - n - 12;          // tile1 column pixel = w0+12+n
    if (dx1 >= -4 && dx1 <= 4) {
      const int s = wv * 9 + (dx1 + 4);
      out[obase + (size_t)s * Hn * Wn] = c1[r];
    }
  }
}

// ---------------------------------------------------------------------------
extern "C" void kernel_launch(void* const* d_in, const int* in_sizes, int n_in,
                              void* d_out, int out_size, void* d_ws, size_t ws_size,
                              hipStream_t stream) {
  const float* f1 = (const float*)d_in[0];
  const float* f2 = (const float*)d_in[1];

  // Workspace layout:
  //   g1: [B][H][W][C]   bf16 = 134,217,728 B
  //   g2: [B][PH][PW][C] bf16 = 160,432,128 B (zero-padded borders)
  uint16_t* g1 = (uint16_t*)d_ws;
  const size_t g1Elems = (size_t)Bn * Hn * Wn * Cn;
  uint16_t* g2 = g1 + g1Elems;
  const size_t g2Elems = (size_t)Bn * PHn * PWn * Cn;

  // 1) zero padded g2 (borders must be 0)
  const size_t n16 = (g2Elems * sizeof(uint16_t)) / 16;   // uint4 count
  const int zblocks = (int)((n16 + 255) / 256);
  zero_ws_kernel<<<zblocks, 256, 0, stream>>>((uint4*)g2, n16);

  // 2) normalize + compact to bf16
  dim3 gridN(Wn / 32, Hn, 2 * Bn);
  norm_bf16_kernel<<<gridN, 256, 0, stream>>>(f1, f2, g1, g2);

  // 3) banded WMMA correlation
  dim3 gridC(Wn / 16, Hn, Bn);
  corr_wmma_kernel<<<gridC, 288, 0, stream>>>(g1, g2, (float*)d_out);
}
